// SelfAttention_13262859010482
// MI455X (gfx1250) — matrix operand order repaired and checked by reading, not compiled, not run on hardware
//
#include <hip/hip_runtime.h>

// Problem constants (match reference)
constexpr int BB = 4;     // batch
constexpr int CC = 256;   // channels
constexpr int HH = 64;    // attention dim
constexpr int NN = 4096;  // sequence

typedef _Float16 v4h  __attribute__((ext_vector_type(4)));
typedef _Float16 v8h  __attribute__((ext_vector_type(8)));
typedef _Float16 v16h __attribute__((ext_vector_type(16)));
typedef float    v8f  __attribute__((ext_vector_type(8)));

// ---------------------------------------------------------------------------
// WMMA helpers. gfx1250 wave32 layouts (CDNA5 ISA 7.12.2):
//  A (16x32 f16): lane L holds row M=L%16; K-chunks [bk..bk+7] and [bk+16..bk+23]
//                 with bk = 8*(L>=16)  -> two contiguous 16B chunks.
//  B (32x16 f16): lane L holds col N=L%16; same K-chunks as A
//                 ([bk..bk+7] and [bk+16..bk+23], bk = 8*(L>=16)) -> two 16B chunks.
//  C/D (16x16 f32): lane L holds col N=L%16, vgpr r holds row M = r + 8*(L/16).
// ---------------------------------------------------------------------------
__device__ __forceinline__ v8f wmma_f16(v16h a, v16h b, v8f c) {
    return __builtin_amdgcn_wmma_f32_16x16x32_f16(
        /*neg_a=*/false, a, /*neg_b=*/false, b,
        /*c_mod=*/(short)0, c, /*reuse_a=*/false, /*reuse_b=*/false);
}

// base points at (tile_row0, k0) of a K-major tile with row stride `stride` halves
__device__ __forceinline__ v16h load_frag_a(const _Float16* base, int stride, int lane) {
    const _Float16* p = base + (lane & 15) * stride + ((lane >> 4) & 1) * 8;
    v8h lo = *(const v8h*)(p);
    v8h hi = *(const v8h*)(p + 16);
    return __builtin_shufflevector(lo, hi, 0,1,2,3,4,5,6,7,8,9,10,11,12,13,14,15);
}

// base points at (tile_col0, k0) of a K-major (transposed) tile
__device__ __forceinline__ v16h load_frag_b(const _Float16* base, int stride, int lane) {
    const _Float16* p = base + (lane & 15) * stride + ((lane >> 4) & 1) * 8;
    v8h lo = *(const v8h*)(p);
    v8h hi = *(const v8h*)(p + 16);
    return __builtin_shufflevector(lo, hi, 0,1,2,3,4,5,6,7,8,9,10,11,12,13,14,15);
}

// ---------------------------------------------------------------------------
// Kernel 1: 1x1-conv projection  out[b, r, n] = relu(sum_c W[r,c]*X[b,c,n] + bias[r])
// fp32 in -> fp16 out.  64x64 output tile per block, 4 waves, WMMA K=32 steps.
// transposeStore=1 -> store out[b, n, r]  (K-major for later A/B fragment loads)
// ---------------------------------------------------------------------------
__global__ __launch_bounds__(128)
void proj_gemm_relu(const float* __restrict__ W, const float* __restrict__ bias,
                    const float* __restrict__ X, _Float16* __restrict__ out,
                    int OutR, int transposeStore)
{
    const int b    = blockIdx.z;
    const int n0   = blockIdx.x * 64;
    const int r0   = blockIdx.y * 64;
    const int t    = threadIdx.x;
    const int lane = t & 31;
    const int w    = t >> 5;

    __shared__ _Float16 As[64 * 40];   // [r][k], k-contiguous, padded stride 40
    __shared__ _Float16 Bs[64 * 40];   // [n][k] (X tile, K-major)

    v8f acc[2][2] = {};
    const int tR0 = (w >> 1) << 1;     // this wave's 32x32 quadrant
    const int tN0 = (w & 1) << 1;
    const float* Xb = X + (size_t)b * CC * NN;

    for (int kk = 0; kk < CC; kk += 32) {
        // stage W tile (64 rows x 32 k), fp32 -> fp16, k-contiguous b128 stores
        {
            const int row  = t >> 1;
            const int colb = (t & 1) * 16;
            const float* wp = W + (size_t)(r0 + row) * CC + kk + colb;
            v8h h0, h1;
#pragma unroll
            for (int i = 0; i < 8; i += 4) {
                float4 v = *(const float4*)(wp + i);
                h0[i + 0] = (_Float16)v.x; h0[i + 1] = (_Float16)v.y;
                h0[i + 2] = (_Float16)v.z; h0[i + 3] = (_Float16)v.w;
            }
#pragma unroll
            for (int i = 0; i < 8; i += 4) {
                float4 v = *(const float4*)(wp + 8 + i);
                h1[i + 0] = (_Float16)v.x; h1[i + 1] = (_Float16)v.y;
                h1[i + 2] = (_Float16)v.z; h1[i + 3] = (_Float16)v.w;
            }
            *(v8h*)&As[row * 40 + colb]     = h0;
            *(v8h*)&As[row * 40 + colb + 8] = h1;
        }
        // stage X tile (32 k x 64 n) K-major: each thread owns one n column and
        // 16 k values -> 16 lane-coalesced b32 loads, two b128 LDS stores.
        {
            const int n  = t & 63;
            const int kh = (t >> 6) * 16;
            const float* xp = Xb + (size_t)(kk + kh) * NN + n0 + n;
            v8h h0, h1;
#pragma unroll
            for (int i = 0; i < 8; ++i) h0[i] = (_Float16)xp[(size_t)i * NN];
#pragma unroll
            for (int i = 0; i < 8; ++i) h1[i] = (_Float16)xp[(size_t)(i + 8) * NN];
            *(v8h*)&Bs[n * 40 + kh]     = h0;
            *(v8h*)&Bs[n * 40 + kh + 8] = h1;
        }
        __syncthreads();
#pragma unroll
        for (int ri = 0; ri < 2; ++ri) {
            v16h a = load_frag_a(&As[(tR0 + ri) * 16 * 40], 40, lane);
#pragma unroll
            for (int ni = 0; ni < 2; ++ni) {
                v16h bf = load_frag_b(&Bs[(tN0 + ni) * 16 * 40], 40, lane);
                acc[ri][ni] = wmma_f16(a, bf, acc[ri][ni]);
            }
        }
        __syncthreads();
    }

    // bias + ReLU + fp16 store
#pragma unroll
    for (int ri = 0; ri < 2; ++ri) {
#pragma unroll
        for (int ni = 0; ni < 2; ++ni) {
#pragma unroll
            for (int r = 0; r < 8; ++r) {
                const int row = r0 + (tR0 + ri) * 16 + r + ((lane >> 4) << 3);
                const int col = n0 + (tN0 + ni) * 16 + (lane & 15);
                float v = acc[ri][ni][r] + bias[row];
                v = v > 0.f ? v : 0.f;
                const _Float16 hv = (_Float16)v;
                if (transposeStore)
                    out[(size_t)b * NN * OutR + (size_t)col * OutR + row] = hv;
                else
                    out[(size_t)b * OutR * NN + (size_t)row * NN + col] = hv;
            }
        }
    }
}

// ---------------------------------------------------------------------------
// Kernel 2: fused attention for a 32-wide block of output columns m.
//   s[n,m] = sum_h f[h,n] g[h,m]      (WMMA, A = f^T tile, B = g^T tile)
//   beta   = softmax over n            (two-pass; online max/sum in pass 1)
//   o[c,m] = sum_n h[c,n] beta[n,m]    (WMMA, B = P^T from LDS)
//   out    = gamma*o + data
// Block: 128 threads (4 waves). Wave w owns n-strip w*16 (S) and c-strip w*64 (O).
// ---------------------------------------------------------------------------
__global__ __launch_bounds__(128)
void attn_fused(const _Float16* __restrict__ fT,  // [B,N,H]
                const _Float16* __restrict__ gT,  // [B,N,H]
                const _Float16* __restrict__ hV,  // [B,C,N]
                const float* __restrict__ data,   // [B,C,N]
                const float* __restrict__ gamma_p,
                float* __restrict__ out)
{
    const int b    = blockIdx.y;
    const int m0   = blockIdx.x * 32;
    const int t    = threadIdx.x;
    const int lane = t & 31;
    const int w    = t >> 5;
    const int mcol = lane & 15;

    __shared__ float   colmax[32];
    __shared__ float   colsum[32];
    __shared__ float   red[4][32];
    __shared__ _Float16 Pt[32 * 72];   // P^T: [m][n], padded stride 72

    const _Float16* fTb = fT + (size_t)b * NN * HH;
    const _Float16* gTb = gT + (size_t)b * NN * HH;
    const _Float16* hb  = hV + (size_t)b * CC * NN;

    if (t < 32) { colmax[t] = -3.0e30f; colsum[t] = 0.f; }
    __syncthreads();

    // ------------------ Pass 1: online column max & exp-sum ------------------
    for (int n0 = 0; n0 < NN; n0 += 64) {
        if (n0 + 64 < NN)
            __builtin_prefetch(fTb + (size_t)(n0 + 64 + w * 16) * HH, 0, 1);

        v8f s[2] = {};
#pragma unroll
        for (int hk = 0; hk < HH; hk += 32) {
            v16h a = load_frag_a(fTb + (size_t)(n0 + w * 16) * HH + hk, HH, lane);
#pragma unroll
            for (int j = 0; j < 2; ++j) {
                v16h q = load_frag_b(gTb + (size_t)(m0 + j * 16) * HH + hk, HH, lane);
                s[j] = wmma_f16(a, q, s[j]);
            }
        }
        // per-column max of this wave's 16 rows
#pragma unroll
        for (int j = 0; j < 2; ++j) {
            float mx = s[j][0];
#pragma unroll
            for (int r = 1; r < 8; ++r) mx = fmaxf(mx, s[j][r]);
            mx = fmaxf(mx, __shfl_xor(mx, 16, 32));
            red[w][j * 16 + mcol] = mx;
        }
        __syncthreads();
        if (t < 32) {
            float bm = fmaxf(fmaxf(red[0][t], red[1][t]), fmaxf(red[2][t], red[3][t]));
            float om = colmax[t];
            float nm = fmaxf(om, bm);
            colsum[t] *= __expf(om - nm);
            colmax[t]  = nm;
        }
        __syncthreads();
        // per-column exp-sum against the updated running max
#pragma unroll
        for (int j = 0; j < 2; ++j) {
            float cm = colmax[j * 16 + mcol];
            float ss = 0.f;
#pragma unroll
            for (int r = 0; r < 8; ++r) ss += __expf(s[j][r] - cm);
            ss += __shfl_xor(ss, 16, 32);
            red[w][j * 16 + mcol] = ss;
        }
        __syncthreads();
        if (t < 32) colsum[t] += red[0][t] + red[1][t] + red[2][t] + red[3][t];
        __syncthreads();
    }
    if (t < 32) colsum[t] = 1.0f / colsum[t];
    __syncthreads();

    // ------------------ Pass 2: P = exp(S-max)/sum, O += h * P ---------------
    v8f o[4][2] = {};
    for (int n0 = 0; n0 < NN; n0 += 64) {
        if (n0 + 64 < NN)
            __builtin_prefetch(hb + (size_t)(w * 64) * NN + n0 + 64, 0, 1);

        v8f s[2] = {};
#pragma unroll
        for (int hk = 0; hk < HH; hk += 32) {
            v16h a = load_frag_a(fTb + (size_t)(n0 + w * 16) * HH + hk, HH, lane);
#pragma unroll
            for (int j = 0; j < 2; ++j) {
                v16h q = load_frag_b(gTb + (size_t)(m0 + j * 16) * HH + hk, HH, lane);
                s[j] = wmma_f16(a, q, s[j]);
            }
        }
        // normalized probabilities -> LDS as P^T (m-major, n-contiguous)
#pragma unroll
        for (int j = 0; j < 2; ++j) {
            const float cm  = colmax[j * 16 + mcol];
            const float inv = colsum[j * 16 + mcol];
            v8h hv;
#pragma unroll
            for (int r = 0; r < 8; ++r)
                hv[r] = (_Float16)(__expf(s[j][r] - cm) * inv);
            *(v8h*)&Pt[(j * 16 + mcol) * 72 + w * 16 + ((lane >> 4) << 3)] = hv;
        }
        __syncthreads();
        // O accumulation over this n-block: K = 64 in two WMMA steps
#pragma unroll
        for (int kb = 0; kb < 64; kb += 32) {
            v16h p0 = load_frag_b(&Pt[0 * 16 * 72 + kb], 72, lane);
            v16h p1 = load_frag_b(&Pt[16 * 72 + kb], 72, lane);
#pragma unroll
            for (int ci = 0; ci < 4; ++ci) {
                v16h ha = load_frag_a(hb + (size_t)(w * 64 + ci * 16) * NN + n0 + kb, NN, lane);
                o[ci][0] = wmma_f16(ha, p0, o[ci][0]);
                o[ci][1] = wmma_f16(ha, p1, o[ci][1]);
            }
        }
        __syncthreads();
    }

    // ------------------ Epilogue: out = gamma*o + data -----------------------
    const float gamma  = gamma_p[0];
    const float* datab = data + (size_t)b * CC * NN;
    float* outb        = out + (size_t)b * CC * NN;
#pragma unroll
    for (int ci = 0; ci < 4; ++ci) {
#pragma unroll
        for (int j = 0; j < 2; ++j) {
#pragma unroll
            for (int r = 0; r < 8; ++r) {
                const int c = w * 64 + ci * 16 + r + ((lane >> 4) << 3);
                const int m = m0 + j * 16 + mcol;
                const size_t idx = (size_t)c * NN + m;
                outb[idx] = gamma * o[ci][j][r] + datab[idx];
            }
        }
    }
}

// ---------------------------------------------------------------------------
// Launch
// ---------------------------------------------------------------------------
extern "C" void kernel_launch(void* const* d_in, const int* in_sizes, int n_in,
                              void* d_out, int out_size, void* d_ws, size_t ws_size,
                              hipStream_t stream)
{
    const float* data  = (const float*)d_in[0];
    const float* Wf    = (const float*)d_in[1];
    const float* bf    = (const float*)d_in[2];
    const float* Wg    = (const float*)d_in[3];
    const float* bg    = (const float*)d_in[4];
    const float* Wh    = (const float*)d_in[5];
    const float* bh    = (const float*)d_in[6];
    const float* gamma = (const float*)d_in[7];

    // workspace: fT [B,N,H] f16 | gT [B,N,H] f16 | h [B,C,N] f16  (~12 MB)
    _Float16* fT = (_Float16*)d_ws;
    _Float16* gT = fT + (size_t)BB * NN * HH;
    _Float16* hV = gT + (size_t)BB * NN * HH;

    dim3 blk(128);
    proj_gemm_relu<<<dim3(NN / 64, HH / 64, BB), blk, 0, stream>>>(Wf, bf, data, fT, HH, 1);
    proj_gemm_relu<<<dim3(NN / 64, HH / 64, BB), blk, 0, stream>>>(Wg, bg, data, gT, HH, 1);
    proj_gemm_relu<<<dim3(NN / 64, CC / 64, BB), blk, 0, stream>>>(Wh, bh, data, hV, CC, 0);
    attn_fused<<<dim3(NN / 32, BB), blk, 0, stream>>>(fT, gT, hV, data, gamma, (float*)d_out);
}
